// SG_56255481643116
// MI455X (gfx1250) — matrix operand order repaired and checked by reading, not compiled, not run on hardware
//
#include <hip/hip_runtime.h>
#include <math.h>

#define B_ 8
#define N_ 1024
#define KNN_ 20
#define F3_ 128
#define INVS 0.9999950000374997f /* 1/sqrt(1+1e-5) */

typedef __attribute__((ext_vector_type(16))) __bf16 v16bf;
typedef __attribute__((ext_vector_type(8)))  float  v8f;

union BF16x16 {
    v16bf v;
    unsigned short u[16];
    unsigned int   w[8];   // dword j = elements (2j, 2j+1)
};

__device__ __forceinline__ unsigned short f2bf(float f) {
    union { float f; unsigned int u; } x; x.f = f;
    unsigned int r = x.u + 0x7FFFu + ((x.u >> 16) & 1u);
    return (unsigned short)(r >> 16);
}

__device__ __forceinline__ v8f wmma_bf(const BF16x16& a, const BF16x16& b, v8f c) {
    return __builtin_amdgcn_wmma_f32_16x16x32_bf16(false, a.v, false, b.v,
                                                   (short)0, c, false, false);
}

// split (B,131,N) -> (B,N,C) f32 + bf16, channels [c0, c0+C)
__global__ void k_split(const float* __restrict__ feat, float* __restrict__ xf,
                        unsigned short* __restrict__ xbf, int c0, int C) {
    size_t i = (size_t)blockIdx.x * blockDim.x + threadIdx.x;
    size_t total = (size_t)B_ * N_ * C;
    if (i >= total) return;
    int c = (int)(i % C); size_t bn = i / C;
    int n = (int)(bn % N_); int b = (int)(bn / N_);
    float v = feat[((size_t)b * 131 + c0 + c) * N_ + n];
    xf[i] = v; xbf[i] = f2bf(v);
}

// (B,N,C) f32 -> bf16 with dest row stride/offset (for hcat packing)
__global__ void k_f2bf(const float* __restrict__ src, unsigned short* __restrict__ dst,
                       int C, int dstStride, int dstOff) {
    size_t i = (size_t)blockIdx.x * blockDim.x + threadIdx.x;
    size_t total = (size_t)B_ * N_ * C;
    if (i >= total) return;
    int c = (int)(i % C); size_t bn = i / C;
    dst[bn * dstStride + dstOff + c] = f2bf(src[i]);
}

// KNN: per (b,n) row, top-20 of neg squared distance (self included, ties -> low idx)
template <int C>
__global__ __launch_bounds__(128) void k_knn(const float* __restrict__ x,
                                             int* __restrict__ idx) {
    __shared__ float xn[C];
    __shared__ float dist[N_];
    __shared__ float rv[128];
    __shared__ int   ri[128];
    int tid = threadIdx.x;
    int b = blockIdx.x / N_, n = blockIdx.x % N_;
    const float* xb = x + (size_t)b * N_ * C;
    if (tid < C) xn[tid] = xb[(size_t)n * C + tid];
    __syncthreads();
    float xxn = 0.f;
#pragma unroll
    for (int c = 0; c < C; ++c) xxn += xn[c] * xn[c];
    for (int m = tid; m < N_; m += 128) {
        const float* xm = xb + (size_t)m * C;
        float dot = 0.f, xxm = 0.f;
#pragma unroll
        for (int c = 0; c < C; ++c) { float v = xm[c]; dot += v * xn[c]; xxm += v * v; }
        dist[m] = 2.f * dot - xxn - xxm;
    }
    __syncthreads();
    for (int t = 0; t < KNN_; ++t) {
        float bv = -__builtin_inff(); int bi = N_;
        for (int m = tid; m < N_; m += 128) {
            float v = dist[m];
            if (v > bv) { bv = v; bi = m; }
        }
        rv[tid] = bv; ri[tid] = bi;
        __syncthreads();
        for (int s = 64; s > 0; s >>= 1) {
            if (tid < s) {
                float ov = rv[tid + s]; int oi = ri[tid + s];
                if (ov > rv[tid] || (ov == rv[tid] && oi < ri[tid])) { rv[tid] = ov; ri[tid] = oi; }
            }
            __syncthreads();
        }
        if (tid == 0) { idx[((size_t)b * N_ + n) * KNN_ + t] = ri[0]; dist[ri[0]] = -__builtin_inff(); }
        __syncthreads();
    }
}

// EdgeConv via W1@(nb-ctr)+W2@ctr = W1@nb + (W2-W1)@ctr.
// base = (W2-W1)@ctr once per tile; per k only W1@nb with raw bf16 neighbor rows.
// One wave per 16(O) x 16(N) tile; v_wmma_f32_16x16x32_bf16.
template <int C, int O>
__global__ __launch_bounds__(32) void k_edge_conv(
    const unsigned short* __restrict__ hbf, const int* __restrict__ idx,
    const float* __restrict__ w, const float* __restrict__ gamma,
    const float* __restrict__ beta, float* __restrict__ out) {
    constexpr int Cp  = ((C + 31) / 32) * 32;
    constexpr int ncs = Cp / 32;
    __shared__ int snidx[16 * KNN_];
    __shared__ unsigned int sW1d[16 * Cp / 2];  // W1 (bf16 pairs)
    __shared__ unsigned int sWdd[16 * Cp / 2];  // W2 - W1
    __shared__ unsigned int sctd[16 * Cp / 2];  // center features

    int tid = threadIdx.x;
    int n0 = blockIdx.x * 16, o0 = blockIdx.y * 16, b = blockIdx.z;
    unsigned short* sW1 = (unsigned short*)sW1d;
    unsigned short* sWd = (unsigned short*)sWdd;
    unsigned short* sct = (unsigned short*)sctd;

    for (int i = tid; i < 16 * KNN_; i += 32)
        snidx[i] = idx[((size_t)b * N_ + n0 + i / KNN_) * KNN_ + (i % KNN_)];
    for (int i = tid; i < 16 * Cp; i += 32) {
        int r = i / Cp, c = i % Cp;
        float w1 = 0.f, w2 = 0.f;
        unsigned short ct = 0;
        if (c < C) {
            w1 = w[(size_t)(o0 + r) * (2 * C) + c];
            w2 = w[(size_t)(o0 + r) * (2 * C) + C + c];
            ct = hbf[((size_t)b * N_ + n0 + r) * C + c];
        }
        sW1[i] = f2bf(w1);
        sWd[i] = f2bf(w2 - w1);
        sct[i] = ct;
    }
    __syncthreads();

    const int half = tid >> 4, q = tid & 15;
    const unsigned int* rW1 = sW1d + q * (Cp / 2);
    const unsigned int* rWd = sWdd + q * (Cp / 2);
    const unsigned int* rCt = sctd + q * (Cp / 2);

    // base = (W2-W1) @ ctr  (k-invariant)
    v8f bacc = {};
#pragma unroll
    for (int cs = 0; cs < ncs; ++cs) {
        BF16x16 a, bb;
#pragma unroll
        for (int j = 0; j < 4; ++j) {
            a.w[j]     = rWd[cs * 16 + half * 4 + j];
            a.w[j + 4] = rWd[cs * 16 + 8 + half * 4 + j];
        }
#pragma unroll
        for (int j = 0; j < 4; ++j) {
            bb.w[j]     = rCt[cs * 16 + half * 4 + j];
            bb.w[j + 4] = rCt[cs * 16 + 8 + half * 4 + j];
        }
        bacc = wmma_bf(a, bb, bacc);
    }

    float ga[8], be[8], rm[8];
#pragma unroll
    for (int j = 0; j < 8; ++j) {
        int o = o0 + 8 * half + j;
        ga[j] = gamma[o] * INVS;
        be[j] = beta[o];
        rm[j] = -__builtin_inff();
    }

    for (int k = 0; k < KNN_; ++k) {
        int m = snidx[q * KNN_ + k];
        if (k + 1 < KNN_) {
            int m2 = snidx[q * KNN_ + k + 1];
            __builtin_prefetch(hbf + ((size_t)b * N_ + m2) * C, 0, 1);
        }
        v8f acc = {};
        if constexpr (C == 3) {
            const unsigned short* nbr = hbf + ((size_t)b * N_ + m) * 3;
            BF16x16 a, bb;
#pragma unroll
            for (int j = 0; j < 4; ++j) {
                a.w[j]     = rW1[half * 4 + j];
                a.w[j + 4] = rW1[8 + half * 4 + j];
            }
            unsigned int p0 = (unsigned)nbr[0] | ((unsigned)nbr[1] << 16);
            unsigned int p1 = (unsigned)nbr[2];
            bb.w[0] = half ? 0u : p0;
            bb.w[1] = half ? 0u : p1;
#pragma unroll
            for (int j = 2; j < 8; ++j) bb.w[j] = 0u;
            acc = wmma_bf(a, bb, acc);
        } else {
            const unsigned int* nbrd = (const unsigned int*)(hbf + ((size_t)b * N_ + m) * C);
#pragma unroll
            for (int cs = 0; cs < ncs; ++cs) {
                BF16x16 a, bb;
#pragma unroll
                for (int j = 0; j < 4; ++j) {
                    a.w[j]     = rW1[cs * 16 + half * 4 + j];
                    a.w[j + 4] = rW1[cs * 16 + 8 + half * 4 + j];
                }
#pragma unroll
                for (int j = 0; j < 4; ++j) {
                    bb.w[j]     = nbrd[cs * 16 + half * 4 + j];
                    bb.w[j + 4] = nbrd[cs * 16 + 8 + half * 4 + j];
                }
                acc = wmma_bf(a, bb, acc);
            }
        }
#pragma unroll
        for (int j = 0; j < 8; ++j) {
            float v = ga[j] * (acc[j] + bacc[j]) + be[j];
            v = (v >= 0.f) ? v : 0.2f * v;
            rm[j] = fmaxf(rm[j], v);
        }
    }
    // D: row = 8*half+j (contiguous in O), col = q
    float* orow = out + ((size_t)b * N_ + n0 + q) * O + o0 + 8 * half;
#pragma unroll
    for (int j = 0; j < 8; ++j) orow[j] = rm[j];
}

// Pointwise 1x1 conv (ew): out[b,n,o] = leaky(g*INVS*(W @ x) + b), WMMA bf16
template <int Cin, int O>
__global__ __launch_bounds__(32) void k_pw_conv(
    const unsigned short* __restrict__ hbf, const float* __restrict__ w,
    const float* __restrict__ gamma, const float* __restrict__ beta,
    float* __restrict__ out) {
    constexpr int ncs = Cin / 32;
    __shared__ unsigned int sWd[16 * Cin / 2];
    __shared__ unsigned int sXd[16 * Cin / 2];
    unsigned short* sW = (unsigned short*)sWd;
    unsigned short* sX = (unsigned short*)sXd;
    int tid = threadIdx.x;
    int n0 = blockIdx.x * 16, o0 = blockIdx.y * 16, b = blockIdx.z;
    for (int i = tid; i < 16 * Cin; i += 32) {
        int r = i / Cin, c = i % Cin;
        sW[i] = f2bf(w[(size_t)(o0 + r) * Cin + c]);
        sX[i] = hbf[((size_t)b * N_ + n0 + r) * Cin + c];
    }
    __syncthreads();
    const int half = tid >> 4, q = tid & 15;
    const unsigned int* rW = sWd + q * (Cin / 2);
    const unsigned int* rX = sXd + q * (Cin / 2);
    v8f acc = {};
#pragma unroll
    for (int cs = 0; cs < ncs; ++cs) {
        BF16x16 a, bb;
#pragma unroll
        for (int j = 0; j < 4; ++j) {
            a.w[j]     = rW[cs * 16 + half * 4 + j];
            a.w[j + 4] = rW[cs * 16 + 8 + half * 4 + j];
        }
#pragma unroll
        for (int j = 0; j < 4; ++j) {
            bb.w[j]     = rX[cs * 16 + half * 4 + j];
            bb.w[j + 4] = rX[cs * 16 + 8 + half * 4 + j];
        }
        acc = wmma_bf(a, bb, acc);
    }
    float* orow = out + ((size_t)b * N_ + n0 + q) * O + o0 + 8 * half;
#pragma unroll
    for (int j = 0; j < 8; ++j) {
        int o = o0 + 8 * half + j;
        float v = gamma[o] * INVS * acc[j] + beta[o];
        orow[j] = (v >= 0.f) ? v : 0.2f * v;
    }
}

__global__ __launch_bounds__(128) void k_mean(const float* __restrict__ emb, float* __restrict__ mean) {
    int b = blockIdx.x, f = threadIdx.x;
    float s = 0.f;
    for (int n = 0; n < N_; ++n) s += emb[((size_t)b * N_ + n) * F3_ + f];
    mean[b * F3_ + f] = s * (1.f / N_);
}
__global__ __launch_bounds__(128) void k_gc(const float* __restrict__ mean, const float* __restrict__ attw,
                                            float* __restrict__ gc) {
    int b = blockIdx.x, g = threadIdx.x;
    float s = 0.f;
    for (int f = 0; f < F3_; ++f) s += mean[b * F3_ + f] * attw[f * F3_ + g];
    gc[b * F3_ + g] = tanhf(s);
}
__global__ __launch_bounds__(256) void k_sc(const float* __restrict__ emb, const float* __restrict__ gc,
                                            float* __restrict__ scbuf, float* __restrict__ att_out) {
    int i = blockIdx.x * 256 + threadIdx.x;  // over B*N
    int b = i / N_;
    const float* er = emb + (size_t)i * F3_;
    const float* gr = gc + b * F3_;
    float s = 0.f;
#pragma unroll 8
    for (int f = 0; f < F3_; ++f) s += er[f] * gr[f];
    float sc = 1.f / (1.f + expf(-s));
    scbuf[i] = sc; att_out[i] = sc;
}
__global__ __launch_bounds__(128) void k_pool(const float* __restrict__ emb, const float* __restrict__ scbuf,
                                              float* __restrict__ e) {
    int b = blockIdx.x, f = threadIdx.x;
    float s = 0.f;
    for (int n = 0; n < N_; ++n) s += emb[((size_t)b * N_ + n) * F3_ + f] * scbuf[b * N_ + n];
    e[b * F3_ + f] = s;
}
__global__ __launch_bounds__(128) void k_ntn(const float* __restrict__ e1, const float* __restrict__ e2,
                                             const float* __restrict__ tnw, const float* __restrict__ tnwb,
                                             const float* __restrict__ tnb, float* __restrict__ tsc) {
    __shared__ float ra[128], rb[128];
    int b = blockIdx.x >> 4, t = blockIdx.x & 15, f = threadIdx.x;
    float inner = 0.f;
    for (int g = 0; g < F3_; ++g) inner += tnw[((size_t)f * F3_ + g) * 16 + t] * e2[b * F3_ + g];
    ra[f] = e1[b * F3_ + f] * inner;
    rb[f] = tnwb[t * 256 + f] * e1[b * F3_ + f] + tnwb[t * 256 + 128 + f] * e2[b * F3_ + f];
    __syncthreads();
    for (int s = 64; s > 0; s >>= 1) {
        if (f < s) { ra[f] += ra[f + s]; rb[f] += rb[f + s]; }
        __syncthreads();
    }
    if (f == 0) {
        float v = ra[0] + rb[0] + tnb[t];
        tsc[b * 16 + t] = (v > 0.f) ? v : 0.f;
    }
}
__global__ void k_tail(const float* __restrict__ tsc, const float* __restrict__ fc1w,
                       const float* __restrict__ fc1b, const float* __restrict__ scw,
                       const float* __restrict__ scb, float* __restrict__ score) {
    int b = threadIdx.x;
    if (b >= B_) return;
    float s = scb[0];
    for (int j = 0; j < 16; ++j) {
        float h = fc1b[j];
        for (int t = 0; t < 16; ++t) h += fc1w[j * 16 + t] * tsc[b * 16 + t];
        h = (h > 0.f) ? h : 0.f;
        s += scw[j] * h;
    }
    score[b] = 1.f / (1.f + expf(-s));
}

template <int C, int O>
static void run_edge(const unsigned short* hbf, const float* hf32, const float* w,
                     const float* g, const float* bias, float* outF, int* idxbuf,
                     hipStream_t stream) {
    k_knn<C><<<dim3(B_ * N_), 128, 0, stream>>>(hf32, idxbuf);
    k_edge_conv<C, O><<<dim3(N_ / 16, O / 16, B_), 32, 0, stream>>>(hbf, idxbuf, w, g, bias, outF);
}

extern "C" void kernel_launch(void* const* d_in, const int* in_sizes, int n_in,
                              void* d_out, int out_size, void* d_ws, size_t ws_size,
                              hipStream_t stream) {
    (void)in_sizes; (void)n_in; (void)out_size; (void)ws_size;
    const float* feat1 = (const float*)d_in[0];
    const float* feat2 = (const float*)d_in[1];
    const float *sw1 = (const float*)d_in[2],  *sg1 = (const float*)d_in[3],  *sb1 = (const float*)d_in[4];
    const float *fw1 = (const float*)d_in[5],  *fg1 = (const float*)d_in[6],  *fb1 = (const float*)d_in[7];
    const float *sw2 = (const float*)d_in[8],  *sg2 = (const float*)d_in[9],  *sb2 = (const float*)d_in[10];
    const float *fw2 = (const float*)d_in[11], *fg2 = (const float*)d_in[12], *fb2 = (const float*)d_in[13];
    const float *sw3 = (const float*)d_in[14], *sg3 = (const float*)d_in[15], *sb3 = (const float*)d_in[16];
    const float *fw3 = (const float*)d_in[17], *fg3 = (const float*)d_in[18], *fb3 = (const float*)d_in[19];
    const float *ew  = (const float*)d_in[20], *eg  = (const float*)d_in[21], *eb  = (const float*)d_in[22];
    const float *attw = (const float*)d_in[23];
    const float *tnw  = (const float*)d_in[24], *tnwb = (const float*)d_in[25], *tnb = (const float*)d_in[26];
    const float *fc1w = (const float*)d_in[27], *fc1b = (const float*)d_in[28];
    const float *scw  = (const float*)d_in[29], *scb  = (const float*)d_in[30];

    char* ws = (char*)d_ws;
    size_t off = 0;
    auto alloc = [&](size_t bytes) -> void* {
        void* p = ws + off;
        off += bytes; off = (off + 255) & ~(size_t)255;
        return p;
    };
    float* bufA          = (float*)alloc((size_t)B_ * N_ * 128 * 4);
    float* bufB          = (float*)alloc((size_t)B_ * N_ * 128 * 4);
    unsigned short* bfA  = (unsigned short*)alloc((size_t)B_ * N_ * 128 * 2);
    unsigned short* bfB  = (unsigned short*)alloc((size_t)B_ * N_ * 128 * 2);
    unsigned short* hcat = (unsigned short*)alloc((size_t)B_ * N_ * 256 * 2);
    float* emb           = (float*)alloc((size_t)B_ * N_ * 128 * 4);
    int* idxbuf          = (int*)alloc((size_t)B_ * N_ * KNN_ * 4);
    float* scbuf         = (float*)alloc((size_t)B_ * N_ * 4);
    float* meanb         = (float*)alloc((size_t)B_ * F3_ * 4);
    float* gcb           = (float*)alloc((size_t)B_ * F3_ * 4);
    float* e1            = (float*)alloc((size_t)B_ * F3_ * 4);
    float* e2            = (float*)alloc((size_t)B_ * F3_ * 4);
    float* tscb          = (float*)alloc((size_t)B_ * 16 * 4);

    const int th = 256;
    auto cvt = [&](const float* src, unsigned short* dst, int C, int stride, int offc) {
        size_t tot = (size_t)B_ * N_ * C;
        k_f2bf<<<(tot + th - 1) / th, th, 0, stream>>>(src, dst, C, stride, offc);
    };
    auto pass = [&](const float* feat, float* e_out, float* att_out) {
        size_t tot3 = (size_t)B_ * N_ * 3, tot128 = (size_t)B_ * N_ * 128;
        // xyz branch: 3 -> 64 -> 64 -> 128
        k_split<<<(tot3 + th - 1) / th, th, 0, stream>>>(feat, bufA, bfA, 0, 3);
        run_edge<3, 64>(bfA, bufA, sw1, sg1, sb1, bufB, idxbuf, stream);
        cvt(bufB, bfB, 64, 64, 0);
        run_edge<64, 64>(bfB, bufB, sw2, sg2, sb2, bufA, idxbuf, stream);
        cvt(bufA, bfA, 64, 64, 0);
        run_edge<64, 128>(bfA, bufA, sw3, sg3, sb3, bufB, idxbuf, stream);
        cvt(bufB, hcat, 128, 256, 0);
        // semantic branch: 128 -> 64 -> 64 -> 128
        k_split<<<(tot128 + th - 1) / th, th, 0, stream>>>(feat, bufA, bfA, 3, 128);
        run_edge<128, 64>(bfA, bufA, fw1, fg1, fb1, bufB, idxbuf, stream);
        cvt(bufB, bfB, 64, 64, 0);
        run_edge<64, 64>(bfB, bufB, fw2, fg2, fb2, bufA, idxbuf, stream);
        cvt(bufA, bfA, 64, 64, 0);
        run_edge<64, 128>(bfA, bufA, fw3, fg3, fb3, bufB, idxbuf, stream);
        cvt(bufB, hcat, 128, 256, 128);
        // 1x1 conv on concat (256 -> 128)
        k_pw_conv<256, 128><<<dim3(N_ / 16, 128 / 16, B_), 32, 0, stream>>>(hcat, ew, eg, eb, emb);
        // attention pooling
        k_mean<<<B_, 128, 0, stream>>>(emb, meanb);
        k_gc<<<B_, 128, 0, stream>>>(meanb, attw, gcb);
        k_sc<<<(B_ * N_) / 256, 256, 0, stream>>>(emb, gcb, scbuf, att_out);
        k_pool<<<B_, 128, 0, stream>>>(emb, scbuf, e_out);
    };

    float* out = (float*)d_out;
    pass(feat1, e1, out + B_);                 // att1 at [8, 8200)
    pass(feat2, e2, out + B_ + B_ * N_);       // att2 at [8200, 16392)
    k_ntn<<<B_ * 16, 128, 0, stream>>>(e1, e2, tnw, tnwb, tnb, tscb);
    k_tail<<<1, 32, 0, stream>>>(tscb, fc1w, fc1b, scw, scb, out);  // score at [0,8)
}
